// SheafBuilderOrtho_74509092651431
// MI455X (gfx1250) — matrix operand order repaired and checked by reading, not compiled, not run on hardware
//
#include <hip/hip_runtime.h>

#define DD 6
#define FF 128
#define NNODES 100000
#define NEDGES 20000
#define NNZ 500000
#define TWOF 256
#define HPAD 264            // padded LDS row stride in halves (256 + 8) -> conflict-free b128 reads
#define PSTRIDE 17          // padded p-tile row stride in floats

typedef __attribute__((ext_vector_type(16))) _Float16     v16h;
typedef __attribute__((ext_vector_type(8)))  float        v8f;
typedef __attribute__((ext_vector_type(4)))  unsigned int v4u;

union F16Frag { v16h v; v4u u[2]; };
union H8      { _Float16 h[8]; v4u u; };

// ---------------------------------------------------------------------------
// Kernel 1: mean over the D=6 sub-rows -> (num, 128). One float4 per thread.
// ---------------------------------------------------------------------------
__global__ void __launch_bounds__(256) mean6_kernel(const float* __restrict__ src,
                                                    float* __restrict__ dst, int num) {
  int idx = blockIdx.x * blockDim.x + threadIdx.x;
  if (idx >= num * 32) return;
  int n = idx >> 5, c = idx & 31;
  const float4* s = (const float4*)src;
  float4 acc = make_float4(0.f, 0.f, 0.f, 0.f);
#pragma unroll
  for (int j = 0; j < DD; ++j) {
    float4 t = s[(size_t)(n * DD + j) * 32 + c];
    acc.x += t.x; acc.y += t.y; acc.z += t.z; acc.w += t.w;
  }
  const float inv = 1.0f / 6.0f;
  ((float4*)dst)[(size_t)n * 32 + c] =
      make_float4(acc.x * inv, acc.y * inv, acc.z * inv, acc.w * inv);
}

// ---------------------------------------------------------------------------
// Kernel 2: fused gather + LayerNorm + (h @ W + b) via WMMA + sigmoid +
//           Householder (d=6) + output emission. 64 threads = 2 waves,
//           32 nnz per wave (two 16x16 WMMA C tiles per wave).
// ---------------------------------------------------------------------------
__global__ void __launch_bounds__(64) sheaf_fused_kernel(
    const float* __restrict__ xm, const float* __restrict__ em,
    const int* __restrict__ hidx,
    const float* __restrict__ gamma, const float* __restrict__ beta,
    const float* __restrict__ W, const float* __restrict__ bias,
    int* __restrict__ idx0, int* __restrict__ idx1, float* __restrict__ attr) {
  __shared__ __align__(16) _Float16 sWt[16 * HPAD];       // W^T as f16 (col n contiguous in k)
  __shared__ __align__(16) _Float16 sH[2][32 * HPAD];     // per-wave normalized h rows, f16
  __shared__ float sP[2][32 * PSTRIDE];                   // per-wave 32x16 p tile (transposed C)
  __shared__ int sRow[64], sCol[64];

  const int tid   = threadIdx.x;
  const int wave  = tid >> 5;
  const int lane  = tid & 31;
  const int gbase = blockIdx.x * 64;

  // Stage W^T into LDS as f16; pad column 15 and k>=256 with zeros.
  for (int i = tid; i < 16 * HPAD; i += 64) {
    int n = i / HPAD, k = i - n * HPAD;
    float v = (n < 15 && k < TWOF) ? W[k * 15 + n] : 0.0f;
    sWt[i] = (_Float16)v;
  }
  { int g = gbase + tid;
    sRow[tid] = (g < NNZ) ? hidx[g] : 0;
    sCol[tid] = (g < NNZ) ? hidx[NNZ + g] : 0; }
  __syncthreads();

  // Per-lane 8-column slice: lanes 0-15 cover xm cols 0..127, lanes 16-31 em cols 128..255.
  const int cbase = (lane < 16) ? lane * 8 : 128 + (lane - 16) * 8;
  float g8[8], b8[8];
#pragma unroll
  for (int i = 0; i < 8; ++i) { g8[i] = gamma[cbase + i]; b8[i] = beta[cbase + i]; }

  // Cooperative LayerNorm: one 256-wide row per iteration across the wave.
  for (int r = 0; r < 32; ++r) {
    int slot = wave * 32 + r;
    const float* src = (lane < 16)
        ? xm + (size_t)sRow[slot] * FF + lane * 8
        : em + (size_t)sCol[slot] * FF + (lane - 16) * 8;
    float4 aq = *(const float4*)src;
    float4 bq = *(const float4*)(src + 4);
    float v[8] = {aq.x, aq.y, aq.z, aq.w, bq.x, bq.y, bq.z, bq.w};
    float s = 0.f, ss = 0.f;
#pragma unroll
    for (int i = 0; i < 8; ++i) { s += v[i]; ss += v[i] * v[i]; }
#pragma unroll
    for (int m = 16; m >= 1; m >>= 1) { s += __shfl_xor(s, m); ss += __shfl_xor(ss, m); }
    float mu   = s * (1.0f / 256.0f);
    float var  = ss * (1.0f / 256.0f) - mu * mu;
    float rstd = rsqrtf(var + 1e-5f);
    H8 hv;
#pragma unroll
    for (int i = 0; i < 8; ++i)
      hv.h[i] = (_Float16)((v[i] - mu) * rstd * g8[i] + b8[i]);
    *(v4u*)&sH[wave][r * HPAD + cbase] = hv.u;   // 16B store, layout-ready for A frags
  }
  __syncthreads();

  // ---- WMMA: two 16x16 C tiles per wave, K=256 in 8 chunks of 32 ----------
  const int ncol = lane & 15;
  float bn = (ncol < 15) ? bias[ncol] : 0.0f;    // fold bias into accumulator
  v8f c0, c1;
#pragma unroll
  for (int j = 0; j < 8; ++j) { c0[j] = bn; c1[j] = bn; }

  const int ko = (lane < 16) ? 0 : 8;            // A frag: lanes 0-15 K0-7/16-23, 16-31 K8-15/24-31
  const int bo = (lane < 16) ? 0 : 8;            // B frag: lanes 0-15 K0-7/16-23, 16-31 K8-15/24-31
#pragma unroll
  for (int kt = 0; kt < 8; ++kt) {
    const int kb = kt * 32;
    F16Frag a0, a1, bf;
    const _Float16* hb = &sH[wave][0];
    a0.u[0] = *(const v4u*)&hb[ ncol       * HPAD + kb + ko];
    a0.u[1] = *(const v4u*)&hb[ ncol       * HPAD + kb + ko + 16];
    a1.u[0] = *(const v4u*)&hb[(ncol + 16) * HPAD + kb + ko];
    a1.u[1] = *(const v4u*)&hb[(ncol + 16) * HPAD + kb + ko + 16];
    bf.u[0] = *(const v4u*)&sWt[ ncol * HPAD + kb + bo];
    bf.u[1] = *(const v4u*)&sWt[ ncol * HPAD + kb + bo + 16];
    c0 = __builtin_amdgcn_wmma_f32_16x16x32_f16(false, a0.v, false, bf.v,
                                                (short)0, c0, false, false);
    c1 = __builtin_amdgcn_wmma_f32_16x16x32_f16(false, a1.v, false, bf.v,
                                                (short)0, c1, false, false);
  }

  // Transpose C tiles through LDS: sP[row][n] = p(row, n). C layout:
  // lanes 0-15 -> M=j, lanes 16-31 -> M=8+j, N = lane%16.
  {
    float* pp = &sP[wave][0];
    const int mb = (lane < 16) ? 0 : 8;
#pragma unroll
    for (int j = 0; j < 8; ++j) {
      pp[(mb + j)      * PSTRIDE + ncol] = c0[j];
      pp[(16 + mb + j) * PSTRIDE + ncol] = c1[j];
    }
  }
  __syncthreads();

  const int g = gbase + tid;
  if (g >= NNZ) return;

  // ---- sigmoid + Householder (d = 6), fully unrolled in registers ---------
  float p[15];
#pragma unroll
  for (int i = 0; i < 15; ++i) {
    float z = sP[wave][lane * PSTRIDE + i];
    p[i] = 1.0f / (1.0f + __expf(-z));
  }

  float A[6][6];
#pragma unroll
  for (int i = 0; i < 6; ++i)
#pragma unroll
    for (int j = 0; j < 6; ++j) A[i][j] = 0.f;
  A[1][0]=p[0];  A[2][0]=p[1];  A[2][1]=p[2];
  A[3][0]=p[3];  A[3][1]=p[4];  A[3][2]=p[5];
  A[4][0]=p[6];  A[4][1]=p[7];  A[4][2]=p[8];  A[4][3]=p[9];
  A[5][0]=p[10]; A[5][1]=p[11]; A[5][2]=p[12]; A[5][3]=p[13]; A[5][4]=p[14];

  float tau[6];
#pragma unroll
  for (int k = 0; k < 6; ++k) {
    float s = 1.0f;
#pragma unroll
    for (int i = 0; i < 6; ++i) s += A[i][k] * A[i][k];
    tau[k] = 2.0f / s;
  }

  float Q[6][6];
#pragma unroll
  for (int i = 0; i < 6; ++i)
#pragma unroll
    for (int j = 0; j < 6; ++j) Q[i][j] = (i == j) ? 1.f : 0.f;

#pragma unroll
  for (int kk = 0; kk < 6; ++kk) {
    const int k = 5 - kk;
    float v[6];
#pragma unroll
    for (int i = 0; i < 6; ++i) v[i] = A[i][k] + ((i == k) ? 1.f : 0.f);
    float w[6];
#pragma unroll
    for (int j = 0; j < 6; ++j) {
      float acc = 0.f;
#pragma unroll
      for (int i = 0; i < 6; ++i) acc += v[i] * Q[i][j];
      w[j] = acc;
    }
#pragma unroll
    for (int i = 0; i < 6; ++i) {
      float tv = tau[k] * v[i];
#pragma unroll
      for (int j = 0; j < 6; ++j) Q[i][j] -= tv * w[j];
    }
  }

  // ---- Emit: idx0 = 6*row + i/6, idx1 = 6*col + i%6, attr = Q row-major ---
  const int r6 = 6 * sRow[tid];
  const int c6 = 6 * sCol[tid];
  const size_t ob = (size_t)g * 36;
  float qf[36];
#pragma unroll
  for (int i = 0; i < 6; ++i)
#pragma unroll
    for (int j = 0; j < 6; ++j) qf[i * 6 + j] = Q[i][j];
#pragma unroll
  for (int q = 0; q < 9; ++q) {
    const int i0 = q * 4;
    int4 o0 = make_int4(r6 + (i0 + 0) / 6, r6 + (i0 + 1) / 6,
                        r6 + (i0 + 2) / 6, r6 + (i0 + 3) / 6);
    int4 o1 = make_int4(c6 + (i0 + 0) % 6, c6 + (i0 + 1) % 6,
                        c6 + (i0 + 2) % 6, c6 + (i0 + 3) % 6);
    float4 oa = make_float4(qf[i0], qf[i0 + 1], qf[i0 + 2], qf[i0 + 3]);
    *(int4*)  (idx0 + ob + i0) = o0;
    *(int4*)  (idx1 + ob + i0) = o1;
    *(float4*)(attr + ob + i0) = oa;
  }
}

// ---------------------------------------------------------------------------
extern "C" void kernel_launch(void* const* d_in, const int* in_sizes, int n_in,
                              void* d_out, int out_size, void* d_ws, size_t ws_size,
                              hipStream_t stream) {
  const float* x     = (const float*)d_in[0];
  const float* e     = (const float*)d_in[1];
  const int*   hidx  = (const int*)d_in[2];
  // d_in[3] node_types, d_in[4] hyperedge_types: unused by the reference math
  const float* gamma = (const float*)d_in[5];
  const float* beta  = (const float*)d_in[6];
  const float* W     = (const float*)d_in[7];
  const float* b     = (const float*)d_in[8];

  float* xm = (float*)d_ws;                          // 100000 x 128 f32
  float* em = xm + (size_t)NNODES * FF;              // 20000  x 128 f32

  int*   idx0 = (int*)d_out;                         // 36*NNZ int32
  int*   idx1 = idx0 + (size_t)NNZ * 36;             // 36*NNZ int32
  float* attr = ((float*)d_out) + 2 * (size_t)NNZ * 36;  // 36*NNZ f32

  { int tot = NNODES * 32;
    mean6_kernel<<<(tot + 255) / 256, 256, 0, stream>>>(x, xm, NNODES); }
  { int tot = NEDGES * 32;
    mean6_kernel<<<(tot + 255) / 256, 256, 0, stream>>>(e, em, NEDGES); }
  { int nb = (NNZ + 63) / 64;
    sheaf_fused_kernel<<<nb, 64, 0, stream>>>(xm, em, hidx, gamma, beta, W, b,
                                              idx0, idx1, attr); }
}
